// SparseConvDecoder_20633022890310
// MI455X (gfx1250) — matrix operand order repaired and checked by reading, not compiled, not run on hardware
//
#include <hip/hip_runtime.h>

typedef __attribute__((ext_vector_type(16))) _Float16 v16h;
typedef __attribute__((ext_vector_type(8)))  _Float16 h8;
typedef __attribute__((ext_vector_type(8)))  float    v8f;

#define A_PAD 40   // halves per LDS row: 80 bytes -> 16B aligned rows

// ---------------------------------------------------------------------------
// Gather(f16, async-to-LDS) -> f16 WMMA GEMM -> scatter-add(atomic f32).
// For kernel offset k = blockIdx.y:
//   Out[out_map[k][i]] += concat(Xa,Xb)[in_map[k][i]] @ W[k]
// Xa/Xb are f16 shadow copies; Wt is f16, pre-transposed to [k][n][kk] with
// kk padded to CinPad (multiple of 32, zero-filled) so B-fragment loads are
// unconditional 16B vector loads from (L2-resident) global memory.
// LDS A-tile double-buffered via GLOBAL_LOAD_ASYNC_TO_LDS_B128 (ASYNCcnt);
// K-loop manually unrolled x2 with two independent B register sets (no
// cross-iteration register copies -> no WMMA hazard NOPs).
// Block: 128 threads = 4 waves. Block tile: 32 gemm rows x 64 cols.
// ---------------------------------------------------------------------------
__global__ void __launch_bounds__(128)
spconv_gemm_f16(const _Float16* __restrict__ Xa, int Ca,
                const _Float16* __restrict__ Xb, int Cb,
                const _Float16* __restrict__ Wt, int CinPad,
                const int* __restrict__ in_map,
                const int* __restrict__ out_map,
                float* __restrict__ Out,
                int n, int Cin, int Cout)
{
    __shared__ __align__(16) _Float16 As[2][32][A_PAD];  // 32 rows x 32 K
    __shared__ int rowIn[32];
    __shared__ int rowOut[32];

    const int tid   = threadIdx.x;
    const int lane  = tid & 31;
    const int wv    = tid >> 5;
    const int k     = blockIdx.y;
    const int nbBlk = blockIdx.z * 64;

    if (tid < 32) {
        const int i = blockIdx.x * 32 + tid;
        rowIn [tid] = in_map [k * n + i];
        rowOut[tid] = out_map[k * n + i];
    }
    __syncthreads();

    const int  myNb   = nbBlk + wv * 16;
    const bool active = (myNb < Cout);          // wave-uniform

    // A staging coordinates: one 16B group per thread per chunk
    const int ar   = tid >> 2;                  // row 0..31
    const int ac0  = (tid & 3) << 3;            // col group 0,8,16,24
    const int aRow = rowIn[ar];

    const int nChunks = (Cin + 31) >> 5;

    // B fragment source: Wt[k][ncol][kk], row length CinPad (zero-padded)
    const _Float16* wrow =
        Wt + (unsigned)(k * Cout + myNb + (lane & 15)) * (unsigned)CinPad;
    const int bb = (lane < 16) ? 0 : 8;         // K 0..7,16..23 | 8..15,24..31

    v8f acc0 = {};
    v8f acc1 = {};

    // async gather of one K-chunk into LDS buffer p (no VGPR round-trip)
    auto stageA = [&](int ch, int p) {
        const int cg = (ch << 5) + ac0;         // 16B group: never straddles
        const unsigned lds = (unsigned)(size_t)&As[p][ar][ac0];
        if (cg < Cin) {                         // Ca/Cin are multiples of 16
            const _Float16* src = (cg < Ca)
                ? (Xa + (unsigned)aRow * Ca + cg)
                : (Xb + (unsigned)aRow * Cb + (cg - Ca));
            asm volatile("global_load_async_to_lds_b128 %0, %1, off"
                         :: "v"(lds), "v"((unsigned long long)(size_t)src)
                         : "memory");
        } else {                                // zero tail (Cin % 32 == 16)
            h8 z = {};
            *(h8*)&As[p][ar][ac0] = z;
        }
    };

    auto loadB = [&](int ch) -> v16h {
        const int k0 = ch << 5;
        h8 blo = *(const h8*)(wrow + k0 + bb);
        h8 bhi = *(const h8*)(wrow + k0 + bb + 16);
        v16h b;
        #pragma unroll
        for (int i = 0; i < 8; ++i) { b[i] = blo[i]; b[i + 8] = bhi[i]; }
        return b;
    };

    // one K-chunk of math from LDS buffer p with B fragment bf
    auto compute = [&](int p, const v16h& bf) {
        const int m  = lane & 15;
        const int ab = (lane < 16) ? 0 : 8;
        h8 alo0 = *(const h8*)&As[p][m][ab];
        h8 ahi0 = *(const h8*)&As[p][m][ab + 16];
        h8 alo1 = *(const h8*)&As[p][16 + m][ab];
        h8 ahi1 = *(const h8*)&As[p][16 + m][ab + 16];
        v16h a0, a1;
        #pragma unroll
        for (int i = 0; i < 8; ++i) {
            a0[i] = alo0[i]; a0[i + 8] = ahi0[i];
            a1[i] = alo1[i]; a1[i + 8] = ahi1[i];
        }
        acc0 = __builtin_amdgcn_wmma_f32_16x16x32_f16(
                   false, a0, false, bf, (short)0, acc0, false, false);
        acc1 = __builtin_amdgcn_wmma_f32_16x16x32_f16(
                   false, a1, false, bf, (short)0, acc1, false, false);
    };

    v16h b0 = {}, b1 = {};
    stageA(0, 0);
    if (active) b0 = loadB(0);
    asm volatile("s_wait_asynccnt 0x0" ::: "memory");
    __syncthreads();

    for (int ch = 0; ch < nChunks; ch += 2) {
        const bool m1 = (ch + 1 < nChunks);
        if (m1) {                                // prefetch chunk ch+1
            stageA(ch + 1, 1);
            if (active) b1 = loadB(ch + 1);
        }
        if (active) compute(0, b0);              // math on chunk ch
        asm volatile("s_wait_asynccnt 0x0" ::: "memory");
        __syncthreads();

        if (m1) {
            const bool m2 = (ch + 2 < nChunks);
            if (m2) {                            // prefetch chunk ch+2
                stageA(ch + 2, 0);
                if (active) b0 = loadB(ch + 2);
            }
            if (active) compute(1, b1);          // math on chunk ch+1
            asm volatile("s_wait_asynccnt 0x0" ::: "memory");
            __syncthreads();
        }
    }

    if (active) {
        // D layout: elem r -> row m = r + (lane>=16 ? 8 : 0), col = lane&15
        const unsigned ncol  = (unsigned)(myNb + (lane & 15));
        const int      mbase = (lane >> 4) << 3;
        #pragma unroll
        for (int r = 0; r < 8; ++r) {
            const unsigned o0 = (unsigned)rowOut[mbase + r] * (unsigned)Cout + ncol;
            __hip_atomic_fetch_add(&Out[o0], acc0[r],
                                   __ATOMIC_RELAXED, __HIP_MEMORY_SCOPE_AGENT);
        }
        #pragma unroll
        for (int r = 0; r < 8; ++r) {
            const unsigned o1 = (unsigned)rowOut[16 + mbase + r] * (unsigned)Cout + ncol;
            __hip_atomic_fetch_add(&Out[o1], acc1[r],
                                   __ATOMIC_RELAXED, __HIP_MEMORY_SCOPE_AGENT);
        }
    }
}

// ---------------------------------------------------------------------------
__global__ void fill0(float* __restrict__ p, long nelem)
{
    const long stride = (long)gridDim.x * blockDim.x;
    for (long i = (long)blockIdx.x * blockDim.x + threadIdx.x; i < nelem; i += stride)
        p[i] = 0.f;
}

__global__ void cvt_f16(const float* __restrict__ src, _Float16* __restrict__ dst,
                        long nelem)
{
    const long stride = (long)gridDim.x * blockDim.x;
    for (long i = (long)blockIdx.x * blockDim.x + threadIdx.x; i < nelem; i += stride)
        dst[i] = (_Float16)src[i];
}

// Wt[k][n][kkp] = (f16) W[k][kkp][n]  (kkp < Cin), 0 for Cin <= kkp < CinPad
__global__ void wt_prep(const float* __restrict__ W, _Float16* __restrict__ Wt,
                        int K, int Cin, int CinPad, int Cout)
{
    const long total  = (long)K * Cout * CinPad;
    const long stride = (long)gridDim.x * blockDim.x;
    for (long i = (long)blockIdx.x * blockDim.x + threadIdx.x; i < total; i += stride) {
        const int kk = (int)(i % CinPad);
        const int nk = (int)(i / CinPad);
        const int nn = nk % Cout;
        const int k  = nk / Cout;
        float v = 0.f;
        if (kk < Cin) v = W[((unsigned)(k * Cin + kk)) * Cout + nn];
        Wt[i] = (_Float16)v;
    }
}

// blockDim.x == C ; per-channel sum / sumsq with grid-stride over rows
__global__ void bn_reduce(const float* __restrict__ X, int n, int C,
                          float* __restrict__ stats)
{
    const int c = threadIdx.x;
    float s = 0.f, s2 = 0.f;
    for (int r = blockIdx.x; r < n; r += gridDim.x) {
        const float v = X[(unsigned)r * C + c];
        s += v; s2 += v * v;
    }
    __hip_atomic_fetch_add(&stats[c],     s,  __ATOMIC_RELAXED, __HIP_MEMORY_SCOPE_AGENT);
    __hip_atomic_fetch_add(&stats[C + c], s2, __ATOMIC_RELAXED, __HIP_MEMORY_SCOPE_AGENT);
}

// Normalize + ReLU in place; optionally emit f16 shadow for next level's gather
__global__ void bn_apply(float* __restrict__ X, int n, int C,
                         const float* __restrict__ stats,
                         const float* __restrict__ g,
                         const float* __restrict__ bta,
                         _Float16* __restrict__ xh)
{
    const long total  = (long)n * C;
    const long stride = (long)gridDim.x * blockDim.x;
    const float inv_n = 1.f / (float)n;
    for (long i = (long)blockIdx.x * blockDim.x + threadIdx.x; i < total; i += stride) {
        const int   c    = (int)(i % C);
        const float mean = stats[c] * inv_n;
        const float var  = stats[C + c] * inv_n - mean * mean;
        float v = (X[i] - mean) * rsqrtf(var + 1e-5f) * g[c] + bta[c];
        v = v > 0.f ? v : 0.f;
        X[i] = v;
        if (xh) xh[i] = (_Float16)v;
    }
}

// ---------------------------------------------------------------------------
extern "C" void kernel_launch(void* const* d_in, const int* in_sizes, int n_in,
                              void* d_out, int out_size, void* d_ws, size_t ws_size,
                              hipStream_t stream)
{
    static const int NS_[5]  = {262144, 98304, 32768, 12288, 4096};
    static const int ENC_[5] = {32, 64, 96, 128, 256};
    static const int DEC_[5] = {256, 128, 96, 64, 48};

    float* out   = (float*)d_out;
    float* ws    = (float*)d_ws;
    float* y     = ws;                   // up-conv fp32 accumulator (12.58M f32)
    float* stats = ws + 12582912;        // 256 f32

    _Float16* hbase = (_Float16*)(ws + 12583168);
    _Float16* xh    = hbase;             // up-conv input shadow  (max 6.29M h)
    _Float16* yh    = xh + 6291456;      // up-conv output shadow (max 12.58M h)
    _Float16* skiph = yh + 12582912;     // skip shadow           (max 8.39M h)
    _Float16* wth   = skiph + 8388608;   // transposed weights    (max 0.88M h)

    long seg[4];
    {
        long off = 0;
        for (int l = 0; l < 4; ++l) { seg[l] = off; off += (long)NS_[3 - l] * DEC_[l + 1]; }
    }

    for (int l = 0; l < 4; ++l) {
        const int n_in_x = NS_[4 - l];
        const int n_out  = NS_[3 - l];
        const int cin    = DEC_[l];
        const int cout   = DEC_[l + 1];
        const int enc    = ENC_[3 - l];
        const int cinP1  = (cin + 31) & ~31;          // up-conv K padded
        const int cinB   = cout + enc;                // block-conv input channels
        const int cinP2  = (cinB + 31) & ~31;

        const float* skip = (const float*)d_in[3 - l];

        const int base = 5 + 8 * l;
        const float* upW    = (const float*)d_in[base + 0];
        const int*   upIn   = (const int*)  d_in[base + 1];
        const int*   upOut  = (const int*)  d_in[base + 2];
        const float* blkW   = (const float*)d_in[base + 3];
        const int*   blkIn  = (const int*)  d_in[base + 4];
        const int*   blkOut = (const int*)  d_in[base + 5];
        const float* gam    = (const float*)d_in[base + 6];
        const float* bet    = (const float*)d_in[base + 7];

        float* seg_out = out + seg[l];
        const long ylen = (long)n_out * cout;

        // ---- f16 shadow of up-conv input (level 0 only; others come from bn_apply)
        if (l == 0)
            cvt_f16<<<1024, 256, 0, stream>>>((const float*)d_in[4], xh,
                                              (long)n_in_x * cin);

        // ---- transposed up-conv: xh -> y (fp32 atomics) ----
        wt_prep<<<512, 256, 0, stream>>>(upW, wth, 8, cin, cinP1, cout);
        fill0<<<4096, 256, 0, stream>>>(y, ylen);
        {
            dim3 g(n_out / 32, 8, (cout + 63) / 64);
            spconv_gemm_f16<<<g, 128, 0, stream>>>(xh, cin, nullptr, 0, wth, cinP1,
                                                   upIn, upOut, y, n_out, cin, cout);
        }

        // ---- f16 shadows for the block conv gather ----
        cvt_f16<<<2048, 256, 0, stream>>>(y, yh, ylen);
        cvt_f16<<<2048, 256, 0, stream>>>(skip, skiph, (long)n_out * enc);

        // ---- ConvBlock: concat(yh, skiph) -> seg_out ----
        wt_prep<<<512, 256, 0, stream>>>(blkW, wth, 27, cinB, cinP2, cout);
        fill0<<<4096, 256, 0, stream>>>(seg_out, ylen);
        {
            dim3 g(n_out / 32, 27, (cout + 63) / 64);
            spconv_gemm_f16<<<g, 128, 0, stream>>>(yh, cout, skiph, enc, wth, cinP2,
                                                   blkIn, blkOut, seg_out,
                                                   n_out, cinB, cout);
        }

        // ---- BatchNorm (batch stats) + ReLU, in place; emit next xh ----
        fill0<<<1, 256, 0, stream>>>(stats, 2L * cout);
        bn_reduce<<<512, cout, 0, stream>>>(seg_out, n_out, cout, stats);
        bn_apply<<<2048, 256, 0, stream>>>(seg_out, n_out, cout, stats, gam, bet,
                                           (l < 3) ? xh : nullptr);
    }
}
